// ChebyshevBasis_80719615361571
// MI455X (gfx1250) — compile-verified, matrix operand order checked by reading, not run on hardware
//
#include <hip/hip_runtime.h>
#include <hip/hip_bf16.h>
#include <math.h>

// ---------------------------------------------------------------------------
// Problem constants (from reference): B=8192, F_IN=F_OUT=1024, DEGREE=10
// A' = [B, F_IN*12] (x + T0..T10 of tanh(x)),  W' = [F_IN*12, F_OUT]
// out[b,o] = sum_k A'[b,k] * W'[k,o];  kl = 0.5*sum(var + mu^2 - 1 - logvar)
// ---------------------------------------------------------------------------
#define BATCH   8192
#define F_IN    1024
#define F_OUT   1024
#define DP1     11          // DEGREE+1
#define KPI     12          // channels per input feature (x + 11 basis)
#define KTOT    (F_IN * KPI)        // 12288
#define KL_N    ((long)F_IN * F_OUT * DP1)
#define KL_BLOCKS 256

#define BM 256              // block tile M
#define BN 128              // block tile N
#define BK 96               // 8 input features * 12 channels = 3 WMMA k-steps
#define LDS_STRIDE 104      // 96 + 8 pad halves -> 208B rows, 16B aligned
#define AS_HALVES (BM * LDS_STRIDE)             // 26624
#define BS_HALVES (BN * LDS_STRIDE)             // 13312
#define LDS_BYTES ((AS_HALVES + BS_HALVES) * 2) // 79872 B (< 320KB/WGP)

typedef _Float16 h8  __attribute__((ext_vector_type(8)));
typedef _Float16 v16h __attribute__((ext_vector_type(16)));
typedef float    v8f  __attribute__((ext_vector_type(8)));

// ---------------------------------------------------------------------------
// Kernel 1: build W' transposed as WpT[n][k] (K contiguous per column), f16.
// ---------------------------------------------------------------------------
__global__ __launch_bounds__(256)
void prep_weights_kernel(const float* __restrict__ coeff_mean,
                         const float* __restrict__ coeff_log_var,
                         const float* __restrict__ base_weight,
                         const float* __restrict__ eps,
                         _Float16* __restrict__ WpT)
{
    long gid = (long)blockIdx.x * 256 + threadIdx.x;
    if (gid >= (long)KTOT * F_OUT) return;
    int k = (int)(gid % KTOT);
    int n = (int)(gid / KTOT);
    int i = k / KPI;
    int r = k - i * KPI;
    float w;
    if (r == 0) {
        w = base_weight[(size_t)i * F_OUT + n];
    } else {
        size_t idx = ((size_t)i * F_OUT + n) * DP1 + (r - 1);
        w = coeff_mean[idx] + eps[idx] * expf(0.5f * coeff_log_var[idx]);
    }
    WpT[gid] = (_Float16)w;
}

// ---------------------------------------------------------------------------
// Kernel 2/3: deterministic two-stage KL reduction.
// ---------------------------------------------------------------------------
__global__ __launch_bounds__(256)
void kl_partial_kernel(const float* __restrict__ coeff_mean,
                       const float* __restrict__ coeff_log_var,
                       float* __restrict__ partials)
{
    __shared__ float red[256];
    float s = 0.0f;
    for (long i = (long)blockIdx.x * 256 + threadIdx.x; i < KL_N;
         i += (long)KL_BLOCKS * 256) {
        float m = coeff_mean[i];
        float l = coeff_log_var[i];
        s += expf(l) + m * m - 1.0f - l;
    }
    red[threadIdx.x] = s;
    __syncthreads();
    for (int off = 128; off > 0; off >>= 1) {
        if (threadIdx.x < off) red[threadIdx.x] += red[threadIdx.x + off];
        __syncthreads();
    }
    if (threadIdx.x == 0) partials[blockIdx.x] = red[0];
}

__global__ __launch_bounds__(256)
void kl_final_kernel(const float* __restrict__ partials,
                     float* __restrict__ out_kl)
{
    __shared__ float red[256];
    red[threadIdx.x] = partials[threadIdx.x];
    __syncthreads();
    for (int off = 128; off > 0; off >>= 1) {
        if (threadIdx.x < off) red[threadIdx.x] += red[threadIdx.x + off];
        __syncthreads();
    }
    if (threadIdx.x == 0) out_kl[0] = 0.5f * red[0];
}

// ---------------------------------------------------------------------------
// Kernel 4: fused basis-generation + f16 WMMA GEMM.
//   Block = 256 threads (8 waves in a 4x2 grid), tile 256x128x96.
//   Each wave owns a 64x64 strip = 4x4 WMMA tiles (16 wmma per k-step).
//   A tile synthesized in LDS from x (tanh + Chebyshev recurrence).
//   B tile streamed L2 -> LDS with gfx1250 async-to-LDS loads (ASYNCcnt).
// ---------------------------------------------------------------------------
__global__ __launch_bounds__(256, 1)
void fused_gemm_kernel(const float* __restrict__ x,
                       const _Float16* __restrict__ WpT,
                       float* __restrict__ out)
{
    extern __shared__ __align__(16) _Float16 smem[];
    _Float16 (*As)[LDS_STRIDE] = (_Float16(*)[LDS_STRIDE])smem;
    _Float16 (*Bs)[LDS_STRIDE] = (_Float16(*)[LDS_STRIDE])(smem + AS_HALVES);

    const int tid  = threadIdx.x;
    const int lane = tid & 31;
    const int wave = tid >> 5;
    const int wm   = wave >> 1;   // 0..3 -> 64-row strip
    const int wn   = wave & 1;    // 0..1 -> 64-col strip
    const int bm0  = blockIdx.y * BM;
    const int bn0  = blockIdx.x * BN;

    const int lrow  = lane & 15;              // M (or N) index within 16x16 tile
    const int khalf = (lane >> 4) << 3;       // lanes 16..31 start at K=8

    v8f acc[4][4];
#pragma unroll
    for (int mt = 0; mt < 4; ++mt)
#pragma unroll
        for (int nt = 0; nt < 4; ++nt) {
            v8f z = {};
            acc[mt][nt] = z;
        }

    for (int kb = 0; kb < KTOT / BK; ++kb) {
        // ---- B tile: 128 cols x 96 K values, async global->LDS (ASYNCcnt) ----
        const _Float16* wsrc = WpT + (size_t)bn0 * KTOT + (size_t)kb * BK;
#pragma unroll
        for (int j = 0; j < 6; ++j) {
            int c  = tid + j * 256;           // 0..1535 16B chunks
            int nl = c / 12;
            int kc = (c - nl * 12) << 3;
            const _Float16* gsrc = wsrc + (size_t)nl * KTOT + kc;
            unsigned ldst = (unsigned)(size_t)&Bs[nl][kc];
            asm volatile("global_load_async_to_lds_b128 %0, %1, off"
                         :: "v"(ldst), "v"(gsrc) : "memory");
        }
        // ---- A tile: 256 rows x 8 input features, 12 channels each ----
        const int ib0 = kb * 8;   // first input feature of this K chunk
#pragma unroll
        for (int p = 0; p < 8; ++p) {
            int idx = tid + p * 256;          // 0..2047 over (row, feature)
            int row = idx >> 3;
            int f   = idx & 7;
            float xv = x[(size_t)(bm0 + row) * F_IN + ib0 + f];
            float t  = tanhf(xv);
            _Float16* dst = &As[row][f * KPI];
            dst[0] = (_Float16)xv;            // pairs with base_weight row
            dst[1] = (_Float16)1.0f;          // T0
            dst[2] = (_Float16)t;             // T1
            float pm2 = 1.0f, pm1 = t;
#pragma unroll
            for (int d = 2; d <= 10; ++d) {   // T2..T10
                float cur = 2.0f * t * pm1 - pm2;
                dst[1 + d] = (_Float16)cur;
                pm2 = pm1; pm1 = cur;
            }
        }
        asm volatile("s_wait_asynccnt 0x0" ::: "memory");
        __syncthreads();

        // ---- 3 WMMA k-steps of 32, 4x4 tiles per wave ----
#pragma unroll
        for (int ks = 0; ks < 3; ++ks) {
            const int k0 = ks * 32;
            v16h af[4], bf[4];
#pragma unroll
            for (int mt = 0; mt < 4; ++mt) {
                int row = wm * 64 + mt * 16 + lrow;
                h8 lo = *(const h8*)&As[row][k0 + khalf];
                h8 hi = *(const h8*)&As[row][k0 + 16 + khalf];
                af[mt] = __builtin_shufflevector(lo, hi,
                    0,1,2,3,4,5,6,7,8,9,10,11,12,13,14,15);
            }
#pragma unroll
            for (int nt = 0; nt < 4; ++nt) {
                int col = wn * 64 + nt * 16 + lrow;
                h8 lo = *(const h8*)&Bs[col][k0 + khalf];
                h8 hi = *(const h8*)&Bs[col][k0 + 16 + khalf];
                bf[nt] = __builtin_shufflevector(lo, hi,
                    0,1,2,3,4,5,6,7,8,9,10,11,12,13,14,15);
            }
#pragma unroll
            for (int mt = 0; mt < 4; ++mt)
#pragma unroll
                for (int nt = 0; nt < 4; ++nt)
                    acc[mt][nt] = __builtin_amdgcn_wmma_f32_16x16x32_f16(
                        false, af[mt], false, bf[nt],
                        (short)0, acc[mt][nt], false, false);
        }
        __syncthreads();
    }

    // ---- epilogue: C/D layout -> VGPR r holds rows r (lanes 0-15) / r+8 ----
    const int ncol = lane & 15;
    const int mofs = (lane >> 4) * 8;
#pragma unroll
    for (int mt = 0; mt < 4; ++mt)
#pragma unroll
        for (int nt = 0; nt < 4; ++nt) {
            int grow = bm0 + wm * 64 + mt * 16 + mofs;
            int gcol = bn0 + wn * 64 + nt * 16 + ncol;
#pragma unroll
            for (int r = 0; r < 8; ++r)
                out[(size_t)(grow + r) * F_OUT + gcol] = acc[mt][nt][r];
        }
}

// ---------------------------------------------------------------------------
// Host launcher. Inputs: x, coeff_mean, coeff_log_var, base_weight, eps.
// d_out: [8192*1024] poly+linear output followed by 1 KL scalar.
// d_ws : 24 MB f16 W' + 256 partial-sum floats.
// ---------------------------------------------------------------------------
extern "C" void kernel_launch(void* const* d_in, const int* in_sizes, int n_in,
                              void* d_out, int out_size, void* d_ws, size_t ws_size,
                              hipStream_t stream)
{
    const float* x    = (const float*)d_in[0];
    const float* mu   = (const float*)d_in[1];
    const float* lv   = (const float*)d_in[2];
    const float* bw   = (const float*)d_in[3];
    const float* eps  = (const float*)d_in[4];
    float* out = (float*)d_out;

    _Float16* WpT = (_Float16*)d_ws;
    size_t wbytes = ((size_t)KTOT * F_OUT * sizeof(_Float16) + 255) & ~(size_t)255;
    float* partials = (float*)((char*)d_ws + wbytes);

    long wtotal = (long)KTOT * F_OUT;
    prep_weights_kernel<<<(unsigned)((wtotal + 255) / 256), 256, 0, stream>>>(
        mu, lv, bw, eps, WpT);

    kl_partial_kernel<<<KL_BLOCKS, 256, 0, stream>>>(mu, lv, partials);
    kl_final_kernel<<<1, 256, 0, stream>>>(partials,
        out + (size_t)BATCH * F_OUT);

    dim3 grid(F_OUT / BN, BATCH / BM);   // 8 x 32
    fused_gemm_kernel<<<grid, 256, LDS_BYTES, stream>>>(x, WpT, out);
}
